// InterFormerBlock_29678224015869
// MI455X (gfx1250) — matrix operand order repaired and checked by reading, not compiled, not run on hardware
//
#include <hip/hip_runtime.h>
#include <hip/hip_fp16.h>
#include <stdint.h>

// ---------------------------------------------------------------------------
// InterFormer block for MI455X (gfx1250, wave32, WMMA, async global->LDS).
// B=4, T=1024, D=512, H=8, dk=64, DFF=2048.
// ---------------------------------------------------------------------------

typedef __attribute__((ext_vector_type(16))) _Float16 v16h;
typedef __attribute__((ext_vector_type(8)))  float    v8f;

#define DEV static __device__ __forceinline__

union Frag16 { v16h v; uint32_t u[8]; };

// A fragment (16x32 f16, MxK): lane<16 -> M=lane, K in {0..7,16..23};
// lane>=16 -> M=lane-16, K in {8..15,24..31}. Packed as f16 pairs per dword.
DEV v16h load_a_frag(const _Float16* rowp, int k0, int lane) {
  const uint32_t* p = (const uint32_t*)(rowp + k0);
  int kb = (lane & 16) ? 4 : 0;
  Frag16 f;
#pragma unroll
  for (int i = 0; i < 4; ++i) f.u[i] = p[kb + i];
#pragma unroll
  for (int i = 0; i < 4; ++i) f.u[4 + i] = p[8 + kb + i];
  return f.v;
}

// B fragment (32x16 f16, KxN) from N-major storage:
// lane<16 -> N=lane, K in {0..7,16..23}; lane>=16 -> N=lane-16, K in {8..15,24..31}.
DEV v16h load_b_frag(const _Float16* rowp, int k0, int lane) {
  const uint32_t* p = (const uint32_t*)(rowp + k0);
  int kb = (lane & 16) ? 4 : 0;
  Frag16 f;
#pragma unroll
  for (int i = 0; i < 4; ++i) f.u[i] = p[kb + i];
#pragma unroll
  for (int i = 0; i < 4; ++i) f.u[4 + i] = p[8 + kb + i];
  return f.v;
}

DEV void async_load_b128_to_lds(uint32_t lds_off, const void* gptr) {
  uint64_t ga = (uint64_t)(uintptr_t)gptr;
  asm volatile("global_load_async_to_lds_b128 %0, %1, off"
               :: "v"(lds_off), "v"(ga) : "memory");
}
DEV void wait_asynccnt0() { asm volatile("s_wait_asynccnt 0" ::: "memory"); }

// ---------------------------------------------------------------------------
// Elementwise converts
// ---------------------------------------------------------------------------
__global__ void cvt_f32_to_f16(const float* __restrict__ src,
                               _Float16* __restrict__ dst, int n) {
  int i = blockIdx.x * blockDim.x + threadIdx.x;
  if (i < n) dst[i] = (_Float16)src[i];
}

// src [R,C] f32 -> dst [C,R] f16 (N-major weight layout for WMMA B frags)
__global__ void transpose_cvt(const float* __restrict__ src,
                              _Float16* __restrict__ dst, int R, int C) {
  int i = blockIdx.x * blockDim.x + threadIdx.x;
  if (i < R * C) {
    int r = i / C, c = i % C;
    dst[(long)c * R + r] = (_Float16)src[i];
  }
}

// ---------------------------------------------------------------------------
// Tiled WMMA GEMM, 128x128 workgroup tile staged through LDS with
// global_load_async_to_lds_b128 (ASYNCcnt). 8 waves, each owns a 32x64 patch
// (2x4 accumulators = 8 WMMAs per 32-K step). Requires M%128==0, N%128==0,
// K%32==0. B stored N-major (BT[N,K]).
// ---------------------------------------------------------------------------
#define APITCH 40  // 32 K-halves + 8 pad halves (80B row pitch, bank-spread)

__global__ __launch_bounds__(256) void gemm_wmma_tiled(
    const _Float16* __restrict__ A, int lda, long sA,
    const _Float16* __restrict__ BT, int ldb, long sB,
    const float* __restrict__ bias, float* __restrict__ Cf,
    _Float16* __restrict__ Ch, int ldc, long sC, int K, int relu) {
  __shared__ __align__(16) _Float16 smA[128 * APITCH];
  __shared__ __align__(16) _Float16 smB[128 * APITCH];
  int tid = threadIdx.x, lane = tid & 31, wv = tid >> 5;
  int Mbase = blockIdx.y << 7, Nbase = blockIdx.x << 7;
  long ba = (long)blockIdx.z;
  const _Float16* Ag = A + ba * sA;
  const _Float16* Bg = BT + ba * sB;

  // cooperative staging: one b128 per thread per tile
  int lr = tid >> 1;            // 0..127 (tile row)
  int lc = (tid & 1) << 4;      // 0 or 16 halves within the 32-wide K slab
  uint32_t aofs =
      (uint32_t)(uintptr_t)(&smA[0]) + (uint32_t)((lr * APITCH + lc) * 2);
  uint32_t bofs =
      (uint32_t)(uintptr_t)(&smB[0]) + (uint32_t)((lr * APITCH + lc) * 2);

  int wm = (wv >> 1) << 5;  // wave row offset within tile (0/32/64/96)
  int wn = (wv & 1) << 6;   // wave col offset within tile (0/64)

  v8f acc[2][4];
#pragma unroll
  for (int i = 0; i < 2; ++i)
#pragma unroll
    for (int j = 0; j < 4; ++j) acc[i][j] = (v8f){};

  for (int k0 = 0; k0 < K; k0 += 32) {
    if (k0) __syncthreads();  // previous slab fully consumed
    async_load_b128_to_lds(aofs, Ag + (long)(Mbase + lr) * lda + k0 + lc);
    async_load_b128_to_lds(bofs, Bg + (long)(Nbase + lr) * ldb + k0 + lc);
    wait_asynccnt0();
    __syncthreads();          // slab visible to all waves

    v16h af[2], bf[4];
#pragma unroll
    for (int i = 0; i < 2; ++i)
      af[i] = load_a_frag(&smA[(wm + i * 16 + (lane & 15)) * APITCH], 0, lane);
#pragma unroll
    for (int j = 0; j < 4; ++j)
      bf[j] = load_b_frag(&smB[(wn + j * 16 + (lane & 15)) * APITCH], 0, lane);
#pragma unroll
    for (int i = 0; i < 2; ++i)
#pragma unroll
      for (int j = 0; j < 4; ++j)
        acc[i][j] = __builtin_amdgcn_wmma_f32_16x16x32_f16(
            false, af[i], false, bf[j], (short)0, acc[i][j], false, false);
  }

  int colin = lane & 15, rsel = (lane >> 4) << 3;
#pragma unroll
  for (int i = 0; i < 2; ++i) {
    int rowb = Mbase + wm + i * 16 + rsel;
#pragma unroll
    for (int j = 0; j < 4; ++j) {
      int col = Nbase + wn + j * 16 + colin;
      float bv = bias ? bias[col] : 0.f;
#pragma unroll
      for (int r = 0; r < 8; ++r) {
        float v = acc[i][j][r] + bv;
        if (relu) v = fmaxf(v, 0.f);
        long idx = ba * sC + (long)(rowb + r) * ldc + col;
        if (Cf) Cf[idx] = v;
        if (Ch) Ch[idx] = (_Float16)v;
      }
    }
  }
}

// ---------------------------------------------------------------------------
// Simple per-wave-tile WMMA GEMM (for skinny N=64 case only).
// ---------------------------------------------------------------------------
__global__ void gemm_wmma_f16(const _Float16* __restrict__ A, int lda, long sA,
                              const _Float16* __restrict__ BT, int ldb, long sB,
                              const float* __restrict__ bias,
                              float* __restrict__ Cf, _Float16* __restrict__ Ch,
                              int ldc, long sC, int M, int N, int K, int relu) {
  int lane = threadIdx.x & 31;
  int wave = blockIdx.x * (blockDim.x >> 5) + (threadIdx.x >> 5);
  int tilesN = N >> 4;
  int tiles = (M >> 4) * tilesN;
  if (wave >= tiles) return;  // uniform per wave
  int tm = wave / tilesN, tn = wave % tilesN;
  long ba = (long)blockIdx.y;
  const _Float16* Ab = A + ba * sA + (long)(tm * 16 + (lane & 15)) * lda;
  const _Float16* Bb = BT + ba * sB + (long)(tn * 16 + (lane & 15)) * ldb;
  v8f c = {};
  for (int k0 = 0; k0 < K; k0 += 32) {
    v16h a = load_a_frag(Ab, k0, lane);
    v16h b = load_b_frag(Bb, k0, lane);
    c = __builtin_amdgcn_wmma_f32_16x16x32_f16(false, a, false, b, (short)0, c,
                                               false, false);
  }
  int row0 = tm * 16 + ((lane >> 4) << 3);
  int col = tn * 16 + (lane & 15);
  float bv = bias ? bias[col] : 0.f;
#pragma unroll
  for (int r = 0; r < 8; ++r) {
    float v = c[r] + bv;
    if (relu) v = fmaxf(v, 0.f);
    long idx = ba * sC + (long)(row0 + r) * ldc + col;
    if (Cf) Cf[idx] = v;
    if (Ch) Ch[idx] = (_Float16)v;
  }
}

// ---------------------------------------------------------------------------
// V_combined: mean over heads, stored transposed: VcT16[b][dc][t]
// ---------------------------------------------------------------------------
__global__ void vc_kernel(const float* __restrict__ V,
                          _Float16* __restrict__ VcT16) {
  int i = blockIdx.x * blockDim.x + threadIdx.x;
  if (i >= 4 * 1024 * 64) return;
  int dc = i & 63;
  int t = (i >> 6) & 1023;
  int b = i >> 16;
  const float* vp = V + ((long)(b * 1024 + t) << 9) + dc;
  float s = 0.f;
#pragma unroll
  for (int h = 0; h < 8; ++h) s += vp[h * 64];
  VcT16[((long)(b * 64 + dc) << 10) + t] = (_Float16)(s * 0.125f);
}

// ---------------------------------------------------------------------------
// Fused attention: per block = (b, 16 query rows). For each head: WMMA scores
// -> LDS, causal mask, rowmax, 16 concurrent bitonic sorts, wave32-scan exact
// 1.5-entmax tau, accumulate weights/H.  ~128KB dynamic LDS (of 320KB/WGP).
// ---------------------------------------------------------------------------
__global__ void attn_entmax_kernel(const _Float16* __restrict__ Q16,
                                   const _Float16* __restrict__ K16,
                                   float* __restrict__ wavg,
                                   _Float16* __restrict__ wavg16) {
  extern __shared__ float sm[];
  float* zbuf = sm;                // 16*1024
  float* sbuf = sm + 16 * 1024;    // 16*1024
  float* rowmax = sm + 32 * 1024;  // 16
  float* taus = rowmax + 16;       // 16
  int tid = threadIdx.x, lane = tid & 31, wv = tid >> 5;
  int b = blockIdx.x >> 6;
  int q0 = (blockIdx.x & 63) << 4;

  for (int h = 0; h < 8; ++h) {
    const _Float16* Qb =
        Q16 + ((long)(b * 1024 + q0 + (lane & 15)) << 9) + h * 64;
    v16h a0 = load_a_frag(Qb, 0, lane);
    v16h a1 = load_a_frag(Qb, 32, lane);
    for (int kt = wv; kt < 64; kt += 8) {
      const _Float16* Kb =
          K16 + ((long)(b * 1024 + kt * 16 + (lane & 15)) << 9) + h * 64;
      v16h b0 = load_b_frag(Kb, 0, lane);
      v16h b1 = load_b_frag(Kb, 32, lane);
      v8f c = {};
      c = __builtin_amdgcn_wmma_f32_16x16x32_f16(false, a0, false, b0, (short)0,
                                                 c, false, false);
      c = __builtin_amdgcn_wmma_f32_16x16x32_f16(false, a1, false, b1, (short)0,
                                                 c, false, false);
      int row0 = (lane >> 4) << 3;
      int col = kt * 16 + (lane & 15);
#pragma unroll
      for (int r = 0; r < 8; ++r) {
        int row = row0 + r;
        float v = (col <= q0 + row) ? c[r] * 0.0625f : -5.0e8f;
        zbuf[(row << 10) + col] = v;
      }
    }
    __syncthreads();

    for (int rr = 0; rr < 2; ++rr) {
      int r = wv * 2 + rr;
      float m = -3.4e38f;
      for (int c2 = lane; c2 < 1024; c2 += 32)
        m = fmaxf(m, zbuf[(r << 10) + c2]);
#pragma unroll
      for (int o = 16; o > 0; o >>= 1) m = fmaxf(m, __shfl_xor(m, o, 32));
      if (lane == 0) rowmax[r] = m;
    }
    __syncthreads();

    for (int i = tid; i < 16 * 1024; i += 256) {
      float z = zbuf[i] - rowmax[i >> 10];
      zbuf[i] = z;
      sbuf[i] = z;
    }
    __syncthreads();

    for (int k = 2; k <= 1024; k <<= 1)
      for (int j = k >> 1; j > 0; j >>= 1) {
        for (int i = tid; i < 16 * 1024; i += 256) {
          int r = i >> 10, x = i & 1023, l = x ^ j;
          if (l > x) {
            float va = sbuf[(r << 10) + x], vb = sbuf[(r << 10) + l];
            bool asc = (x & k) != 0;
            if (asc ? (va > vb) : (va < vb)) {
              sbuf[(r << 10) + x] = vb;
              sbuf[(r << 10) + l] = va;
            }
          }
        }
        __syncthreads();
      }

    for (int rr = 0; rr < 2; ++rr) {
      int r = wv * 2 + rr;
      float c1 = 0.f, c2 = 0.f, best = 0.f;
      for (int ch = 0; ch < 32; ++ch) {
        int idx = (ch << 5) + lane;
        float v = sbuf[(r << 10) + idx];
        float s1 = v, s2 = v * v;
#pragma unroll
        for (int o = 1; o < 32; o <<= 1) {
          float t1 = __shfl_up(s1, o, 32);
          float t2 = __shfl_up(s2, o, 32);
          if (lane >= o) { s1 += t1; s2 += t2; }
        }
        s1 += c1;
        s2 += c2;
        float kk = (float)(idx + 1);
        float mean = s1 / kk, msq = s2 / kk;
        float ssv = kk * (msq - mean * mean);
        float delta = (1.f - ssv) / kk;
        float tau = mean - sqrtf(fmaxf(delta, 0.f));
        unsigned long long msk = __ballot(tau <= v);
        if (msk) {
          int hi = 63 - __builtin_clzll(msk);
          best = __shfl(tau, hi, 32);
        }
        c1 = __shfl(s1, 31, 32);
        c2 = __shfl(s2, 31, 32);
      }
      if (lane == 0) taus[r] = best;
    }
    __syncthreads();

    for (int i = tid; i < 16 * 1024; i += 256) {
      int r = i >> 10, col = i & 1023;
      float d = fmaxf(zbuf[i] - taus[r], 0.f);
      float w = d * d * 0.125f;
      long gi = ((long)(b * 1024 + q0 + r) << 10) + col;
      if (h == 0) {
        wavg[gi] = w;
      } else {
        float acc = wavg[gi] + w;
        wavg[gi] = acc;
        if (h == 7) wavg16[gi] = (_Float16)acc;
      }
    }
    __syncthreads();
  }
}

// ---------------------------------------------------------------------------
// LayerNorm(A + B) * g + beta over rows of 512; optional f16 copy.
// ---------------------------------------------------------------------------
__global__ void ln_kernel(const float* __restrict__ A,
                          const float* __restrict__ Bres,
                          const float* __restrict__ g,
                          const float* __restrict__ be,
                          float* __restrict__ outf, _Float16* __restrict__ outh) {
  __shared__ float red[256];
  __shared__ float s_mu, s_rs;
  int row = blockIdx.x, tid = threadIdx.x;
  const float* a = A + ((long)row << 9);
  const float* br = Bres + ((long)row << 9);
  float v0 = a[tid] + br[tid];
  float v1 = a[tid + 256] + br[tid + 256];
  red[tid] = v0 + v1;
  __syncthreads();
  for (int o = 128; o > 0; o >>= 1) {
    if (tid < o) red[tid] += red[tid + o];
    __syncthreads();
  }
  if (tid == 0) s_mu = red[0] * (1.f / 512.f);
  __syncthreads();
  float mu = s_mu;
  float d0 = v0 - mu, d1 = v1 - mu;
  red[tid] = d0 * d0 + d1 * d1;
  __syncthreads();
  for (int o = 128; o > 0; o >>= 1) {
    if (tid < o) red[tid] += red[tid + o];
    __syncthreads();
  }
  if (tid == 0) s_rs = rsqrtf(red[0] * (1.f / 512.f) + 1e-5f);
  __syncthreads();
  float rs = s_rs;
  float o0 = d0 * rs * g[tid] + be[tid];
  float o1 = d1 * rs * g[tid + 256] + be[tid + 256];
  long base = (long)row << 9;
  if (outf) { outf[base + tid] = o0; outf[base + tid + 256] = o1; }
  if (outh) {
    outh[base + tid] = (_Float16)o0;
    outh[base + tid + 256] = (_Float16)o1;
  }
}

// ---------------------------------------------------------------------------
// Host side
// ---------------------------------------------------------------------------
static void launch_gemm_tiled(const _Float16* A, int lda, long sA,
                              const _Float16* BT, int ldb, long sB,
                              const float* bias, float* Cf, _Float16* Ch,
                              int ldc, long sC, int M, int N, int K, int relu,
                              int batch, hipStream_t stream) {
  dim3 g(N / 128, M / 128, batch);
  gemm_wmma_tiled<<<g, 256, 0, stream>>>(A, lda, sA, BT, ldb, sB, bias, Cf, Ch,
                                         ldc, sC, K, relu);
}

static void launch_gemm(const _Float16* A, int lda, long sA, const _Float16* BT,
                        int ldb, long sB, const float* bias, float* Cf,
                        _Float16* Ch, int ldc, long sC, int M, int N, int K,
                        int relu, int batch, hipStream_t stream) {
  int tiles = (M / 16) * (N / 16);
  dim3 g((tiles + 7) / 8, batch, 1);
  gemm_wmma_f16<<<g, 256, 0, stream>>>(A, lda, sA, BT, ldb, sB, bias, Cf, Ch,
                                       ldc, sC, M, N, K, relu);
}

extern "C" void kernel_launch(void* const* d_in, const int* in_sizes, int n_in,
                              void* d_out, int out_size, void* d_ws,
                              size_t ws_size, hipStream_t stream) {
  (void)in_sizes; (void)n_in; (void)out_size; (void)ws_size;
  const float* x  = (const float*)d_in[0];
  const float* wq = (const float*)d_in[1];
  const float* bq = (const float*)d_in[2];
  const float* wk = (const float*)d_in[3];
  const float* bk = (const float*)d_in[4];
  const float* wv = (const float*)d_in[5];
  const float* bv = (const float*)d_in[6];
  const float* wo = (const float*)d_in[7];
  const float* bo = (const float*)d_in[8];
  const float* w1 = (const float*)d_in[9];
  const float* b1 = (const float*)d_in[10];
  const float* w2 = (const float*)d_in[11];
  const float* b2 = (const float*)d_in[12];
  const float* ga = (const float*)d_in[13];
  const float* ba = (const float*)d_in[14];
  const float* gf = (const float*)d_in[15];
  const float* bf = (const float*)d_in[16];

  const long Bn = 4, Tn = 1024, Dm = 512, DKn = 64, DFF = 2048;
  const long Mr = Bn * Tn;  // 4096

  char* wsb = (char*)d_ws;
  size_t off = 0;
  auto take = [&](size_t bytes) -> char* {
    char* p = wsb + off;
    off = (off + bytes + 255) & ~(size_t)255;
    return p;
  };
  _Float16* x16   = (_Float16*)take(Mr * Dm * 2);
  _Float16* wqT   = (_Float16*)take(Dm * Dm * 2);
  _Float16* wkT   = (_Float16*)take(Dm * Dm * 2);
  _Float16* wvT   = (_Float16*)take(Dm * Dm * 2);
  _Float16* woT   = (_Float16*)take(Dm * DKn * 2);
  _Float16* w1T   = (_Float16*)take(DFF * Dm * 2);
  _Float16* w2T   = (_Float16*)take(Dm * DFF * 2);
  _Float16* Q16   = (_Float16*)take(Mr * Dm * 2);
  _Float16* K16   = (_Float16*)take(Mr * Dm * 2);
  float*    Vf    = (float*)take(Mr * Dm * 4);  // reused: attn f32, ffn2 f32
  _Float16* VcT16 = (_Float16*)take(Bn * DKn * Tn * 2);
  _Float16* wavg16 = (_Float16*)take(Bn * Tn * Tn * 2);
  _Float16* Aout16 = (_Float16*)take(Mr * DKn * 2);
  float*    hf    = (float*)take(Mr * Dm * 4);
  _Float16* h16   = (_Float16*)take(Mr * Dm * 2);
  _Float16* f116  = (_Float16*)take(Mr * DFF * 2);

  float* outp = (float*)d_out;   // [4,1024,512]
  float* wavg = outp + Mr * Dm;  // [4,1024,1024]

  // 1) f16 conversions / weight transposes
  cvt_f32_to_f16<<<(Mr * Dm + 255) / 256, 256, 0, stream>>>(x, x16, Mr * Dm);
  transpose_cvt<<<(Dm * Dm + 255) / 256, 256, 0, stream>>>(wq, wqT, Dm, Dm);
  transpose_cvt<<<(Dm * Dm + 255) / 256, 256, 0, stream>>>(wk, wkT, Dm, Dm);
  transpose_cvt<<<(Dm * Dm + 255) / 256, 256, 0, stream>>>(wv, wvT, Dm, Dm);
  transpose_cvt<<<(DKn * Dm + 255) / 256, 256, 0, stream>>>(wo, woT, DKn, Dm);
  transpose_cvt<<<(Dm * DFF + 255) / 256, 256, 0, stream>>>(w1, w1T, Dm, DFF);
  transpose_cvt<<<(DFF * Dm + 255) / 256, 256, 0, stream>>>(w2, w2T, DFF, Dm);

  // 2) QKV projections (tiled WMMA + async LDS staging)
  launch_gemm_tiled(x16, Dm, 0, wqT, Dm, 0, bq, nullptr, Q16, Dm, 0, Mr, Dm, Dm,
                    0, 1, stream);
  launch_gemm_tiled(x16, Dm, 0, wkT, Dm, 0, bk, nullptr, K16, Dm, 0, Mr, Dm, Dm,
                    0, 1, stream);
  launch_gemm_tiled(x16, Dm, 0, wvT, Dm, 0, bv, Vf, nullptr, Dm, 0, Mr, Dm, Dm,
                    0, 1, stream);

  // 3) V_combined (head mean), transposed f16
  vc_kernel<<<(Bn * Tn * DKn + 255) / 256, 256, 0, stream>>>(Vf, VcT16);

  // 4) fused WMMA-scores + exact 1.5-entmax + head-average
  size_t shbytes = (size_t)(16 * 1024 * 2 + 32) * sizeof(float);
  attn_entmax_kernel<<<dim3(Bn * (Tn / 16)), 256, shbytes, stream>>>(
      Q16, K16, wavg, wavg16);

  // 5) attn_out = (weights_avg @ V_combined) @ wo + bo
  launch_gemm(wavg16, Tn, Tn * Tn, VcT16, Tn, DKn * Tn, nullptr, nullptr,
              Aout16, DKn, Tn * DKn, Tn, DKn, Tn, 0, Bn, stream);
  float* attnf = Vf;  // reuse (V no longer needed)
  launch_gemm_tiled(Aout16, DKn, 0, woT, DKn, 0, bo, attnf, nullptr, Dm, 0, Mr,
                    Dm, DKn, 0, 1, stream);

  // 6) h = LN(x + attn_out)
  ln_kernel<<<Mr, 256, 0, stream>>>(x, attnf, ga, ba, hf, h16);

  // 7) FFN: relu(h@w1+b1)@w2+b2
  launch_gemm_tiled(h16, Dm, 0, w1T, Dm, 0, b1, nullptr, f116, DFF, 0, Mr, DFF,
                    Dm, 1, 1, stream);
  float* ffn2 = Vf;  // reuse again (attnf consumed by ln1)
  launch_gemm_tiled(f116, DFF, 0, w2T, DFF, 0, b2, ffn2, nullptr, Dm, 0, Mr, Dm,
                    DFF, 0, 1, stream);

  // 8) out = LN(ffn + h)
  ln_kernel<<<Mr, 256, 0, stream>>>(ffn2, hf, gf, bf, outp, nullptr);
}
